// MultiHeadAttention_1202590843013
// MI455X (gfx1250) — matrix operand order repaired and checked by reading, not compiled, not run on hardware
//
#include <hip/hip_runtime.h>
#include <hip/hip_bf16.h>

// ---------------- vector types ----------------
typedef __attribute__((ext_vector_type(16))) __bf16 v16bf;
typedef __attribute__((ext_vector_type(8)))  __bf16 v8bf;
typedef __attribute__((ext_vector_type(4)))  __bf16 v4bf;
typedef __attribute__((ext_vector_type(8)))  float  v8f;
typedef __attribute__((ext_vector_type(4)))  float  v4f;

// Problem constants
#define BB 2
#define SS 2048
#define DD 1024
#define HH 16
#define HD 64
#define MM (BB*SS)                      // 4096 tokens
#define OUT_ELEMS ((size_t)BB*SS*DD)    // 4,194,304 (att region follows in d_out)

// ---------------- fragment loaders ----------------
// A-matrix 16x32 bf16 (ISA 7.12.2): lane row = lane&15; K chunks {k0+g..+7} and
// {k0+16+g..+7} with g = 8*(lane>>4)
__device__ __forceinline__ v16bf ldA_bf16(const __bf16* row, int k0, int g) {
    v8bf lo = *(const v8bf*)(row + k0 + g);
    v8bf hi = *(const v8bf*)(row + k0 + 16 + g);
    v16bf r;
#pragma unroll
    for (int i = 0; i < 8; ++i) { r[i] = lo[i]; r[i + 8] = hi[i]; }
    return r;
}

// B-matrix 32x16 bf16: lane col = lane&15; K chunks {k0+g..+7} and {k0+16+g..+7} with g = 8*(lane>>4)
__device__ __forceinline__ v16bf ldB_bf16(const __bf16* row, int k0, int half) {
    const __bf16* p = row + k0 + half * 8;
    v8bf lo = *(const v8bf*)(p);
    v8bf hi = *(const v8bf*)(p + 16);
    v16bf r;
#pragma unroll
    for (int i = 0; i < 8; ++i) { r[i] = lo[i]; r[i + 8] = hi[i]; }
    return r;
}

// A fragment from f32 source (attention probabilities), cvt to bf16 in regs
__device__ __forceinline__ v16bf ldA_f32(const float* row, int k0, int g) {
    v4f a = *(const v4f*)(row + k0 + g);
    v4f b = *(const v4f*)(row + k0 + g + 4);
    v4f c = *(const v4f*)(row + k0 + 16 + g);
    v4f d = *(const v4f*)(row + k0 + 16 + g + 4);
    v16bf r;
#pragma unroll
    for (int i = 0; i < 4; ++i) {
        r[i]      = (__bf16)a[i];
        r[4 + i]  = (__bf16)b[i];
        r[8 + i]  = (__bf16)c[i];
        r[12 + i] = (__bf16)d[i];
    }
    return r;
}

// ---------------- elementwise f32 -> bf16 ----------------
__global__ void cvt_f32_to_bf16(const float* __restrict__ src,
                                __bf16* __restrict__ dst, int n4) {
    int i = blockIdx.x * blockDim.x + threadIdx.x;
    if (i < n4) {
        v4f x = ((const v4f*)src)[i];
        v4bf y;
#pragma unroll
        for (int j = 0; j < 4; ++j) y[j] = (__bf16)x[j];
        ((v4bf*)dst)[i] = y;
    }
}

// ---------------- GEMM: C[4096x1024] = A[4096x1024] * W^T + bias ----------------
// Wave computes 16x64: A fragment reused for 4 WMMAs; k-loop software-pipelined
// (next iteration's 5 fragments loaded before current WMMAs so waits are partial).
// MODE 0: bf16 head layout [b][h][s][d]  (Q,K)
// MODE 1: bf16 head-T layout [b][h][d][s] (V transposed)
// MODE 2: f32 row-major [m][n]            (output projection)
template <int MODE>
__global__ void gemm_proj(const __bf16* __restrict__ A,
                          const __bf16* __restrict__ W,
                          const float* __restrict__ bias,
                          void* __restrict__ dst) {
    const int K = DD;
    int lane = threadIdx.x & 31, wid = threadIdx.x >> 5;
    int t   = blockIdx.x * 8 + wid;          // wave-tile id, < 256*16
    int ng  = t & 15;                        // 64-wide N group
    int mt  = t >> 4;                        // 16-row M tile
    int nl = lane & 15, half = lane >> 4, g = half * 8;

    const __bf16* aRow  = A + (size_t)(mt * 16 + nl) * K;
    const __bf16* wRow0 = W + (size_t)(ng * 64 +  0 + nl) * K;
    const __bf16* wRow1 = W + (size_t)(ng * 64 + 16 + nl) * K;
    const __bf16* wRow2 = W + (size_t)(ng * 64 + 32 + nl) * K;
    const __bf16* wRow3 = W + (size_t)(ng * 64 + 48 + nl) * K;

    v8f c0 = {}, c1 = {}, c2 = {}, c3 = {};
    // prologue loads
    v16bf af = ldA_bf16(aRow, 0, g);
    v16bf b0 = ldB_bf16(wRow0, 0, half);
    v16bf b1 = ldB_bf16(wRow1, 0, half);
    v16bf b2 = ldB_bf16(wRow2, 0, half);
    v16bf b3 = ldB_bf16(wRow3, 0, half);
    for (int k0 = 0; k0 < K; k0 += 32) {
        int kn = (k0 + 32 < K) ? (k0 + 32) : k0;   // last iter: harmless reload
        v16bf afn = ldA_bf16(aRow, kn, g);
        v16bf b0n = ldB_bf16(wRow0, kn, half);
        v16bf b1n = ldB_bf16(wRow1, kn, half);
        v16bf b2n = ldB_bf16(wRow2, kn, half);
        v16bf b3n = ldB_bf16(wRow3, kn, half);
        c0 = __builtin_amdgcn_wmma_f32_16x16x32_bf16(false, af, false, b0, (short)0, c0, false, false);
        c1 = __builtin_amdgcn_wmma_f32_16x16x32_bf16(false, af, false, b1, (short)0, c1, false, false);
        c2 = __builtin_amdgcn_wmma_f32_16x16x32_bf16(false, af, false, b2, (short)0, c2, false, false);
        c3 = __builtin_amdgcn_wmma_f32_16x16x32_bf16(false, af, false, b3, (short)0, c3, false, false);
        af = afn; b0 = b0n; b1 = b1n; b2 = b2n; b3 = b3n;
    }

    int m0 = mt * 16 + half * 8;
    v8f acc[4] = {c0, c1, c2, c3};
#pragma unroll
    for (int j = 0; j < 4; ++j) {
        int n = ng * 64 + j * 16 + nl;
        float bn = bias[n];
        if (MODE == 2) {
            float* out = (float*)dst;
#pragma unroll
            for (int i = 0; i < 8; ++i)
                out[(size_t)(m0 + i) * DD + n] = acc[j][i] + bn;
        } else {
            __bf16* o = (__bf16*)dst;
            int h = n >> 6, d = n & 63;
#pragma unroll
            for (int i = 0; i < 8; ++i) {
                int m = m0 + i;
                int b = m >> 11, s = m & 2047;
                float v = acc[j][i] + bn;
                size_t idx = (MODE == 0)
                    ? ((((size_t)b * HH + h) * SS + s) * HD + d)
                    : ((((size_t)b * HH + h) * HD + d) * SS + s);
                o[idx] = (__bf16)v;
            }
        }
    }
}

// ---------------- scores: att_raw = scale * (Q K^T), causal mask ----------------
// Wave computes 16 q-rows x 64 k-cols (4 tiles), Q fragment reused 4x.
__global__ void scores_kernel(const __bf16* __restrict__ Qh,
                              const __bf16* __restrict__ Kh,
                              float* __restrict__ att) {
    const float scale = 0.02209708691f;  // 1/sqrt(2048)
    const float NEG = -__builtin_inff();
    int lane = threadIdx.x & 31, wid = threadIdx.x >> 5;
    int t   = blockIdx.x * 8 + wid;          // < 32*128*32
    int kg  = t & 31;                        // 64-wide k group
    int qt  = (t >> 5) & 127;
    int bh  = t >> 12;
    int nl = lane & 15, half = lane >> 4, g = half * 8;

    int mrow0 = qt * 16 + half * 8;

    if (kg * 64 > qt * 16 + 15) {            // whole 16x64 group above diagonal
#pragma unroll
        for (int j = 0; j < 4; ++j) {
            int kk = kg * 64 + j * 16 + nl;
#pragma unroll
            for (int i = 0; i < 8; ++i)
                att[((size_t)bh * SS + (mrow0 + i)) * SS + kk] = NEG;
        }
        return;
    }

    const __bf16* qRow  = Qh + ((size_t)bh * SS + qt * 16 + nl) * HD;
    const __bf16* kRow0 = Kh + ((size_t)bh * SS + kg * 64 +  0 + nl) * HD;
    const __bf16* kRow1 = Kh + ((size_t)bh * SS + kg * 64 + 16 + nl) * HD;
    const __bf16* kRow2 = Kh + ((size_t)bh * SS + kg * 64 + 32 + nl) * HD;
    const __bf16* kRow3 = Kh + ((size_t)bh * SS + kg * 64 + 48 + nl) * HD;

    v8f c0 = {}, c1 = {}, c2 = {}, c3 = {};
#pragma unroll
    for (int k0 = 0; k0 < HD; k0 += 32) {
        v16bf af = ldA_bf16(qRow, k0, g);
        v16bf b0 = ldB_bf16(kRow0, k0, half);
        v16bf b1 = ldB_bf16(kRow1, k0, half);
        v16bf b2 = ldB_bf16(kRow2, k0, half);
        v16bf b3 = ldB_bf16(kRow3, k0, half);
        c0 = __builtin_amdgcn_wmma_f32_16x16x32_bf16(false, af, false, b0, (short)0, c0, false, false);
        c1 = __builtin_amdgcn_wmma_f32_16x16x32_bf16(false, af, false, b1, (short)0, c1, false, false);
        c2 = __builtin_amdgcn_wmma_f32_16x16x32_bf16(false, af, false, b2, (short)0, c2, false, false);
        c3 = __builtin_amdgcn_wmma_f32_16x16x32_bf16(false, af, false, b3, (short)0, c3, false, false);
    }

    v8f acc[4] = {c0, c1, c2, c3};
#pragma unroll
    for (int j = 0; j < 4; ++j) {
        int kk = kg * 64 + j * 16 + nl;
#pragma unroll
        for (int i = 0; i < 8; ++i) {
            int q = mrow0 + i;
            float v = (kk > q) ? NEG : acc[j][i] * scale;
            att[((size_t)bh * SS + q) * SS + kk] = v;
        }
    }
}

// ---------------- row softmax over att (2048-wide rows), in place ----------------
__global__ void softmax_rows(float* __restrict__ att) {
    __shared__ float red[8];
    size_t row = blockIdx.x;                 // 65536 rows
    float* p = att + row * SS;
    int tid = threadIdx.x, lane = tid & 31, wid = tid >> 5;

    float v[8];
#pragma unroll
    for (int j = 0; j < 8; ++j) v[j] = p[j * 256 + tid];

    float m = -__builtin_inff();
#pragma unroll
    for (int j = 0; j < 8; ++j) m = fmaxf(m, v[j]);
#pragma unroll
    for (int off = 16; off > 0; off >>= 1) m = fmaxf(m, __shfl_xor(m, off, 32));
    if (lane == 0) red[wid] = m;
    __syncthreads();
    float bm = red[0];
#pragma unroll
    for (int k = 1; k < 8; ++k) bm = fmaxf(bm, red[k]);
    __syncthreads();

    float e[8], s = 0.f;
#pragma unroll
    for (int j = 0; j < 8; ++j) { e[j] = __expf(v[j] - bm); s += e[j]; }
#pragma unroll
    for (int off = 16; off > 0; off >>= 1) s += __shfl_xor(s, off, 32);
    if (lane == 0) red[wid] = s;
    __syncthreads();
    float bs = 0.f;
#pragma unroll
    for (int k = 0; k < 8; ++k) bs += red[k];
    float inv = 1.0f / bs;
#pragma unroll
    for (int j = 0; j < 8; ++j) p[j * 256 + tid] = e[j] * inv;
}

// ---------------- AV: O = att @ V ----------------
// Wave computes 16 q-rows x full head dim (64): the f32 att fragment is loaded
// ONCE per k-step (att read exactly once overall) and the k-loop is software-
// pipelined to overlap next-step loads with current WMMAs.
__global__ void av_kernel(const float* __restrict__ att,
                          const __bf16* __restrict__ Vt,
                          __bf16* __restrict__ O) {
    int lane = threadIdx.x & 31, wid = threadIdx.x >> 5;
    int t  = blockIdx.x * 8 + wid;           // < 32*128
    int qt = t & 127;
    int bh = t >> 7;
    int nl = lane & 15, half = lane >> 4, g = half * 8;

    const float*  aRow  = att + ((size_t)bh * SS + qt * 16 + nl) * SS;
    const __bf16* vRow0 = Vt + ((size_t)bh * HD +  0 + nl) * SS;
    const __bf16* vRow1 = Vt + ((size_t)bh * HD + 16 + nl) * SS;
    const __bf16* vRow2 = Vt + ((size_t)bh * HD + 32 + nl) * SS;
    const __bf16* vRow3 = Vt + ((size_t)bh * HD + 48 + nl) * SS;

    v8f c0 = {}, c1 = {}, c2 = {}, c3 = {};
    v16bf af = ldA_f32(aRow, 0, g);
    v16bf b0 = ldB_bf16(vRow0, 0, half);
    v16bf b1 = ldB_bf16(vRow1, 0, half);
    v16bf b2 = ldB_bf16(vRow2, 0, half);
    v16bf b3 = ldB_bf16(vRow3, 0, half);
    for (int k0 = 0; k0 < SS; k0 += 32) {
        int kn = (k0 + 32 < SS) ? (k0 + 32) : k0;
        v16bf afn = ldA_f32(aRow, kn, g);
        v16bf b0n = ldB_bf16(vRow0, kn, half);
        v16bf b1n = ldB_bf16(vRow1, kn, half);
        v16bf b2n = ldB_bf16(vRow2, kn, half);
        v16bf b3n = ldB_bf16(vRow3, kn, half);
        c0 = __builtin_amdgcn_wmma_f32_16x16x32_bf16(false, af, false, b0, (short)0, c0, false, false);
        c1 = __builtin_amdgcn_wmma_f32_16x16x32_bf16(false, af, false, b1, (short)0, c1, false, false);
        c2 = __builtin_amdgcn_wmma_f32_16x16x32_bf16(false, af, false, b2, (short)0, c2, false, false);
        c3 = __builtin_amdgcn_wmma_f32_16x16x32_bf16(false, af, false, b3, (short)0, c3, false, false);
        af = afn; b0 = b0n; b1 = b1n; b2 = b2n; b3 = b3n;
    }

    int b = bh >> 4, h = bh & 15;
    int s0 = qt * 16 + half * 8;
    v8f acc[4] = {c0, c1, c2, c3};
#pragma unroll
    for (int j = 0; j < 4; ++j) {
        int n = h * HD + j * 16 + nl;
#pragma unroll
        for (int i = 0; i < 8; ++i) {
            int m = b * SS + s0 + i;
            O[(size_t)m * DD + n] = (__bf16)acc[j][i];
        }
    }
}

// ---------------- host-side launch ----------------
extern "C" void kernel_launch(void* const* d_in, const int* in_sizes, int n_in,
                              void* d_out, int out_size, void* d_ws, size_t ws_size,
                              hipStream_t stream) {
    (void)in_sizes; (void)n_in; (void)out_size; (void)ws_size;

    const float* query = (const float*)d_in[0];
    const float* key_  = (const float*)d_in[1];
    const float* value = (const float*)d_in[2];
    const float* Wq = (const float*)d_in[3];
    const float* bq = (const float*)d_in[4];
    const float* Wk = (const float*)d_in[5];
    const float* bk = (const float*)d_in[6];
    const float* Wv = (const float*)d_in[7];
    const float* bv = (const float*)d_in[8];
    const float* Wo = (const float*)d_in[9];
    const float* bo = (const float*)d_in[10];

    // workspace carve-out (bf16)
    char* ws = (char*)d_ws;
    size_t o = 0;
    const size_t ACT = (size_t)MM * DD;      // 4M elems
    const size_t WEL = (size_t)DD * DD;      // 1M elems
    __bf16* Xq  = (__bf16*)(ws + o); o += ACT * 2;
    __bf16* Xk  = (__bf16*)(ws + o); o += ACT * 2;
    __bf16* Xv  = (__bf16*)(ws + o); o += ACT * 2;
    __bf16* Wqb = (__bf16*)(ws + o); o += WEL * 2;
    __bf16* Wkb = (__bf16*)(ws + o); o += WEL * 2;
    __bf16* Wvb = (__bf16*)(ws + o); o += WEL * 2;
    __bf16* Wob = (__bf16*)(ws + o); o += WEL * 2;
    __bf16* Qh  = (__bf16*)(ws + o); o += ACT * 2;
    __bf16* Kh  = (__bf16*)(ws + o); o += ACT * 2;
    __bf16* Vt  = (__bf16*)(ws + o); o += ACT * 2;
    __bf16* Obf = (__bf16*)(ws + o); o += ACT * 2;

    float* outp = (float*)d_out;
    float* attp = (float*)d_out + OUT_ELEMS;

    // 1) conversions to bf16
    cvt_f32_to_bf16<<<(int)(ACT / 4 / 256), 256, 0, stream>>>(query, Xq, (int)(ACT / 4));
    cvt_f32_to_bf16<<<(int)(ACT / 4 / 256), 256, 0, stream>>>(key_,  Xk, (int)(ACT / 4));
    cvt_f32_to_bf16<<<(int)(ACT / 4 / 256), 256, 0, stream>>>(value, Xv, (int)(ACT / 4));
    cvt_f32_to_bf16<<<(int)(WEL / 4 / 256), 256, 0, stream>>>(Wq, Wqb, (int)(WEL / 4));
    cvt_f32_to_bf16<<<(int)(WEL / 4 / 256), 256, 0, stream>>>(Wk, Wkb, (int)(WEL / 4));
    cvt_f32_to_bf16<<<(int)(WEL / 4 / 256), 256, 0, stream>>>(Wv, Wvb, (int)(WEL / 4));
    cvt_f32_to_bf16<<<(int)(WEL / 4 / 256), 256, 0, stream>>>(Wo, Wob, (int)(WEL / 4));

    // 2) projections: 256 M-tiles x 16 N-groups = 4096 wave-tiles / 8 waves
    const int gemmBlocks = (256 * 16) / 8;
    gemm_proj<0><<<gemmBlocks, 256, 0, stream>>>(Xq, Wqb, bq, (void*)Qh);
    gemm_proj<0><<<gemmBlocks, 256, 0, stream>>>(Xk, Wkb, bk, (void*)Kh);
    gemm_proj<1><<<gemmBlocks, 256, 0, stream>>>(Xv, Wvb, bv, (void*)Vt);

    // 3) scores: 32 bh * 128 qt * 32 kgroups
    scores_kernel<<<(32 * 128 * 32) / 8, 256, 0, stream>>>(Qh, Kh, attp);

    // 4) softmax: one block per row (32 * 2048 rows)
    softmax_rows<<<32 * 2048, 256, 0, stream>>>(attp);

    // 5) AV: 32 bh * 128 qt wave-tiles (full head dim per wave)
    av_kernel<<<(32 * 128) / 8, 256, 0, stream>>>(attp, Vt, Obf);

    // 6) output projection -> f32 out
    gemm_proj<2><<<gemmBlocks, 256, 0, stream>>>(Obf, Wob, bo, (void*)outp);
}
